// Net_68805376082306
// MI455X (gfx1250) — matrix operand order repaired and checked by reading, not compiled, not run on hardware
//
#include <hip/hip_runtime.h>

typedef __attribute__((ext_vector_type(16))) _Float16 v16h;
typedef __attribute__((ext_vector_type(8)))  float    v8f;

#define WIDTH 64

__device__ __forceinline__ v8f wmma16(v16h a, v16h b, v8f c) {
  // D = A(16x32 f16) * B(32x16 f16) + C(16x16 f32)
  return __builtin_amdgcn_wmma_f32_16x16x32_f16(
      /*neg_a=*/false, a, /*neg_b=*/false, b,
      /*c_mod=*/(short)0, c, /*reuse_a=*/false, /*reuse_b=*/false);
}

// Load one lane's A fragment pair (K=0..63 split in two 16x32 chunks) with
// optional fused ReLU, converting f32 -> f16 (packs to v_cvt_pk_f16_f32).
__device__ __forceinline__ void load_a_frags(const float* __restrict__ hr,
                                             int relu, v16h a[2]) {
#pragma unroll
  for (int c = 0; c < 2; ++c) {
    const float4* q = (const float4*)(hr + c * 32);
    float4 p[4] = {q[0], q[1], q[4], q[5]};
#pragma unroll
    for (int j = 0; j < 4; ++j) {
      float4 v = p[j];
      if (relu) {
        v.x = fmaxf(v.x, 0.f); v.y = fmaxf(v.y, 0.f);
        v.z = fmaxf(v.z, 0.f); v.w = fmaxf(v.w, 0.f);
      }
      a[c][j * 4 + 0] = (_Float16)v.x;
      a[c][j * 4 + 1] = (_Float16)v.y;
      a[c][j * 4 + 2] = (_Float16)v.z;
      a[c][j * 4 + 3] = (_Float16)v.w;
    }
  }
}

// ---------------- degree / normalization ----------------
__global__ void k_deg_init(float* deg, int n) {
  int i = blockIdx.x * blockDim.x + threadIdx.x;
  if (i < n) deg[i] = 1.0f;  // self-loop
}

__global__ void k_deg_count(const int* __restrict__ ei, float* deg, int E) {
  int e = blockIdx.x * blockDim.x + threadIdx.x;
  if (e < E) unsafeAtomicAdd(&deg[ei[E + e]], 1.0f);  // dst degree (exact fp counts)
}

__global__ void k_dis(float* deg, int n) {
  int i = blockIdx.x * blockDim.x + threadIdx.x;
  if (i < n) deg[i] = rsqrtf(deg[i]);  // in place: deg -> deg^{-1/2}
}

// ---------------- fc1: h = x @ W(3x64) + b ----------------
__global__ void k_fc1(const float* __restrict__ x, const float* __restrict__ w,
                      const float* __restrict__ b, float* __restrict__ h, int n) {
  int t = blockIdx.x * blockDim.x + threadIdx.x;
  if (t >= n * WIDTH) return;
  int i = t >> 6, f = t & 63;
  float x0 = x[i * 3 + 0], x1 = x[i * 3 + 1], x2 = x[i * 3 + 2];
  h[t] = fmaf(x0, w[f], fmaf(x1, w[WIDTH + f], fmaf(x2, w[2 * WIDTH + f], b[f])));
}

// ---------------- WMMA GEMM: HW = relu?(H) @ W(64x64) ----------------
__global__ void __launch_bounds__(256)
k_gemm_wmma(const float* __restrict__ H, const float* __restrict__ W,
            float* __restrict__ HW, int n, int relu) {
  __shared__ _Float16 sW[WIDTH * WIDTH];  // 8 KB f16 weights
  for (int j = threadIdx.x; j < WIDTH * WIDTH; j += 256)
    sW[j] = (_Float16)W[j];
  __syncthreads();

  const int lane = threadIdx.x & 31;
  const int wave = threadIdx.x >> 5;
  const int lo = lane & 15;
  const int up = lane >> 4;  // 0 / 1

  // B fragments: bf[nt][c] covers K = c*32..c*32+31, N = nt*16..nt*16+15
  // lanes 0-15 hold K 0..7,16..23 of the chunk (N = lane), lanes 16-31 hold K 8..15,24..31
  v16h bf[4][2];
#pragma unroll
  for (int nt = 0; nt < 4; ++nt) {
#pragma unroll
    for (int c = 0; c < 2; ++c) {
      const int kb = c * 32 + up * 8;
      const int col = nt * 16 + lo;
#pragma unroll
      for (int i = 0; i < 16; ++i)
        bf[nt][c][i] = sW[(kb + i + (i >= 8 ? 8 : 0)) * WIDTH + col];
    }
  }

  const int ntiles = (n + 15) >> 4;
  const int nw = gridDim.x * 8;
  for (int tile = blockIdx.x * 8 + wave; tile < ntiles; tile += nw) {
    const int m0 = tile << 4;
    const int row = m0 + lo;
    const bool full = (m0 + 16 <= n);  // wave-uniform

    // A fragments: lane row = m0+lo; halves 0..7 = K kb..kb+7, 8..15 = K kb+16..kb+23
    v16h a[2];
    if (full) {
      load_a_frags(H + (size_t)row * WIDTH + up * 8, relu, a);
    } else {
#pragma unroll
      for (int c = 0; c < 2; ++c)
#pragma unroll
        for (int i = 0; i < 16; ++i) a[c][i] = (_Float16)0.0f;
      if (row < n) load_a_frags(H + (size_t)row * WIDTH + up * 8, relu, a);
    }

    v8f acc[4];
#pragma unroll
    for (int nt = 0; nt < 4; ++nt) {
#pragma unroll
      for (int v = 0; v < 8; ++v) acc[nt][v] = 0.0f;
      acc[nt] = wmma16(a[0], bf[nt][0], acc[nt]);
      acc[nt] = wmma16(a[1], bf[nt][1], acc[nt]);
    }

    // C/D layout: VGPR v -> row m0 + v + 8*up, col nt*16 + lo.
    // Fast path: 32 b32 stores at immediate offsets from one base address.
    if (full) {
      float* base = HW + (size_t)(m0 + up * 8) * WIDTH + lo;
#pragma unroll
      for (int v = 0; v < 8; ++v)
#pragma unroll
        for (int nt = 0; nt < 4; ++nt)
          base[v * WIDTH + nt * 16] = acc[nt][v];
    } else {
#pragma unroll
      for (int nt = 0; nt < 4; ++nt)
#pragma unroll
        for (int v = 0; v < 8; ++v) {
          const int r = m0 + v + up * 8;
          if (r < n) HW[(size_t)r * WIDTH + nt * 16 + lo] = acc[nt][v];
        }
    }
  }
}

// ---------------- self-loop + bias init of the accumulator ----------------
__global__ void k_init_accum(const float* __restrict__ hw, const float* __restrict__ dis,
                             const float* __restrict__ bias, float* __restrict__ acc, int n) {
  int t = blockIdx.x * blockDim.x + threadIdx.x;
  if (t >= n * WIDTH) return;
  int i = t >> 6, f = t & 63;
  float s = dis[i];
  acc[t] = fmaf(hw[t], s * s, bias[f]);
}

// ---------------- edge scatter: acc[d] += hw[s] * dis[s]*dis[d] ----------------
__global__ void k_scatter(const int* __restrict__ ei, const float* __restrict__ dis,
                          const float* __restrict__ hw, float* __restrict__ acc, int E) {
  long long t = (long long)blockIdx.x * blockDim.x + threadIdx.x;
  int e = (int)(t >> 4);
  if (e >= E) return;
  int fg = ((int)t & 15) * 4;  // half-wave of 16 lanes covers one 64-f row
  int s = ei[e], d = ei[E + e];
  float w = dis[s] * dis[d];
  float4 v = *(const float4*)(hw + (size_t)s * WIDTH + fg);
  float* p = acc + (size_t)d * WIDTH + fg;
  unsafeAtomicAdd(p + 0, v.x * w);
  unsafeAtomicAdd(p + 1, v.y * w);
  unsafeAtomicAdd(p + 2, v.z * w);
  unsafeAtomicAdd(p + 3, v.w * w);
}

// ---------------- fc2: out = relu(h) @ w(64x1) + b ----------------
__global__ void k_fc2(const float* __restrict__ h, const float* __restrict__ w,
                      const float* __restrict__ b, float* __restrict__ out, int n) {
  int lane = threadIdx.x & 31;
  int node = (int)(((long long)blockIdx.x * blockDim.x + threadIdx.x) >> 5);
  if (node >= n) return;
  int f = lane * 2;
  float2 v = *(const float2*)(h + (size_t)node * WIDTH + f);
  float s = fmaxf(v.x, 0.f) * w[f] + fmaxf(v.y, 0.f) * w[f + 1];
#pragma unroll
  for (int off = 16; off > 0; off >>= 1) s += __shfl_xor(s, off, 32);
  if (lane == 0) out[node] = s + b[0];
}

extern "C" void kernel_launch(void* const* d_in, const int* in_sizes, int n_in,
                              void* d_out, int out_size, void* d_ws, size_t ws_size,
                              hipStream_t stream) {
  const float* x      = (const float*)d_in[0];
  const int*   ei     = (const int*)d_in[1];
  const float* fc1_w  = (const float*)d_in[2];
  const float* fc1_b  = (const float*)d_in[3];
  const float* conv_w = (const float*)d_in[4];
  const float* conv_b = (const float*)d_in[5];
  const float* fc2_w  = (const float*)d_in[6];
  const float* fc2_b  = (const float*)d_in[7];

  const int n = in_sizes[0] / 3;
  const int E = in_sizes[1] / 2;

  char* ws = (char*)d_ws;
  size_t off = 0;
  auto carve = [&](size_t bytes) {
    char* p = ws + off;
    off += (bytes + 255) & ~(size_t)255;
    return p;
  };
  float* deg  = (float*)carve((size_t)n * sizeof(float));  // degrees -> deg^{-1/2}
  float* bufA = (float*)carve((size_t)n * WIDTH * sizeof(float));
  float* bufB = (float*)carve((size_t)n * WIDTH * sizeof(float));
  float* bufC = (float*)carve((size_t)n * WIDTH * sizeof(float));
  (void)ws_size;

  const int T = 256;
  k_deg_init<<<(n + T - 1) / T, T, 0, stream>>>(deg, n);
  k_deg_count<<<(E + T - 1) / T, T, 0, stream>>>(ei, deg, E);
  k_dis<<<(n + T - 1) / T, T, 0, stream>>>(deg, n);

  float* h   = bufA;
  float* hw  = bufB;
  float* acc = bufC;
  k_fc1<<<((size_t)n * WIDTH + T - 1) / T, T, 0, stream>>>(x, fc1_w, fc1_b, h, n);

  const int ntiles = (n + 15) / 16;
  const int gemm_blocks = (ntiles + 7) / 8;
  const long long scat_threads = (long long)E * 16;
  const int scat_blocks = (int)((scat_threads + T - 1) / T);

  for (int L = 0; L < 5; ++L) {
    k_gemm_wmma<<<gemm_blocks, T, 0, stream>>>(h, conv_w + (size_t)L * WIDTH * WIDTH,
                                               hw, n, L > 0 ? 1 : 0);
    k_init_accum<<<((size_t)n * WIDTH + T - 1) / T, T, 0, stream>>>(
        hw, deg, conv_b + (size_t)L * WIDTH, acc, n);
    k_scatter<<<scat_blocks, T, 0, stream>>>(ei, deg, hw, acc, E);
    float* tmp = h; h = acc; acc = hw; hw = tmp;  // ping-pong buffers
  }

  k_fc2<<<((size_t)n * 32 + T - 1) / T, T, 0, stream>>>(h, fc2_w, fc2_b, (float*)d_out, n);
}
